// GCNEncoder_86328842649875
// MI455X (gfx1250) — matrix operand order repaired and checked by reading, not compiled, not run on hardware
//
#include <hip/hip_runtime.h>
#include <cstddef>

// ---------------------------------------------------------------------------
// GraphSAGE-LSTM encoder for MI455X (gfx1250), wave32 + WMMA bf16.
// v3: double-buffered async (ASYNCcnt) staging of W_hh tiles + gathered Xproj
//     pre-activations; staging latency fully overlapped with WMMA compute,
//     one barrier per 16-column tile.
//   * Input projection hoisted: Xproj = feat @ W_ih^T + b_lstm (one big WMMA
//     GEMM per layer, full weight reuse), stored bf16.
//   * Recurrence: h @ W_hh^T via WMMA from LDS-staged weights (shared by all
//     8 waves -> 8x less L2 traffic than private streaming).
// ---------------------------------------------------------------------------

#define NN   32768
#define DEG  16
#define GG   64

#define USE_ASYNC_STAGE 1

typedef __attribute__((ext_vector_type(16))) __bf16 v16bf;
typedef __attribute__((ext_vector_type(8)))  float  v8f;

union FragB16 {
    v16bf v;
    uint4 u[2];
};

static __device__ __forceinline__ unsigned short f32_to_bf16(float f) {
    unsigned int u = __float_as_uint(f);
    u += 0x7FFFu + ((u >> 16) & 1u);   // round-to-nearest-even
    return (unsigned short)(u >> 16);
}
static __device__ __forceinline__ float bf16_to_f32(unsigned short s) {
    return __uint_as_float(((unsigned int)s) << 16);
}
static __device__ __forceinline__ float sigmoidf_fast(float x) {
    return 1.0f / (1.0f + __expf(-x));
}

static __device__ __forceinline__ v8f wmma_bf16(const v16bf a, const v16bf b, v8f c) {
    return __builtin_amdgcn_wmma_f32_16x16x32_bf16(false, a, false, b, (short)0, c, false, false);
}

// 16x32 bf16 A tile from row-major storage with stride AS (elements).
template<int AS>
static __device__ __forceinline__ void load_a_tile(const unsigned short* As,
                                                   int rowBase, int lane, int kbase,
                                                   FragB16& fa) {
    const int m    = lane & 15;
    const int koff = kbase + ((lane >> 4) << 3);
    const unsigned short* p = As + (size_t)(rowBase + m) * AS + koff;
    fa.u[0] = *(const uint4*)(p);
    fa.u[1] = *(const uint4*)(p + 16);
}

// 32x16 bf16 B tile from column-packed weights W[n][k] (row n = output col).
static __device__ __forceinline__ void load_b_tile_s(const unsigned short* W, int stride,
                                                     int colBase, int lane, int kbase,
                                                     FragB16& fb) {
    const int n  = lane & 15;
    const int kb = kbase + ((lane >> 4) << 3);
    const unsigned short* p = W + (size_t)(colBase + n) * stride + kb;
    fb.u[0] = *(const uint4*)(p);
    fb.u[1] = *(const uint4*)(p + 16);
}

// ---- async global -> LDS 16B copy (CDNA5, tracked by ASYNCcnt) ------------
static __device__ __forceinline__ void async_copy_b128(void* lds_dst, const void* gsrc) {
#if USE_ASYNC_STAGE
    unsigned ldsoff = (unsigned)(size_t)lds_dst;
    asm volatile("global_load_async_to_lds_b128 %0, %1, off"
                 :: "v"(ldsoff), "v"((unsigned long long)(size_t)gsrc)
                 : "memory");
#else
    *(uint4*)lds_dst = *(const uint4*)gsrc;
#endif
}
static __device__ __forceinline__ void async_wait() {
#if USE_ASYNC_STAGE
    asm volatile("s_wait_asynccnt 0" ::: "memory");
#endif
}

// ---------------------------------------------------------------------------
// Xproj = featprev @ Wih^T + b   ->  bf16 [N][4D]   (one-time GEMM per layer)
// Grid: N/64 blocks x 128 threads (4 waves x 16 rows).
// ---------------------------------------------------------------------------
template<int D>
__global__ __launch_bounds__(128)
void xproj_kernel(const unsigned short* __restrict__ featprev,
                  const unsigned short* __restrict__ Wih,
                  const float*          __restrict__ bl,
                  unsigned short*       __restrict__ Xproj) {
    const int tid  = threadIdx.x;
    const int lane = tid & 31;
    const int rowBase = (tid >> 5) << 4;
    const int base = blockIdx.x * 64;
    const int m = lane & 15;
    const int koffA = (lane >> 4) << 3;
    const int mhi = (lane >> 4) << 3;

    for (int nt = 0; nt < 4 * D / 16; ++nt) {
        v8f acc = {};
        const int cb = nt * 16;
#pragma unroll
        for (int kk = 0; kk < D / 32; ++kk) {
            const int kbase = kk * 32;
            FragB16 fa, fb;
            const unsigned short* p = featprev + (size_t)(base + rowBase + m) * D + kbase + koffA;
            fa.u[0] = *(const uint4*)(p);
            fa.u[1] = *(const uint4*)(p + 16);
            load_b_tile_s(Wih, D, cb, lane, kbase, fb);
            acc = wmma_bf16(fa.v, fb.v, acc);
        }
        const int ncol = cb + (lane & 15);
        const float bb = bl[ncol];
#pragma unroll
        for (int v = 0; v < 8; ++v) {
            const int node = base + rowBase + v + mhi;
            Xproj[(size_t)node * (4 * D) + ncol] = f32_to_bf16(acc[v] + bb);
        }
    }
}

// ---------------------------------------------------------------------------
// Fused LSTM recurrence + output projection, double-buffered async staging.
// Grid: N/128 blocks x 256 threads (8 waves x 16 rows).
// ---------------------------------------------------------------------------
template<int D, bool RELU_BF16_OUT>
__global__ __launch_bounds__(256)
void sage_lstm_kernel(const unsigned short* __restrict__ Xproj,
                      const unsigned short* __restrict__ featprev,
                      const int*            __restrict__ neighbors,
                      const unsigned short* __restrict__ Whh,
                      const unsigned short* __restrict__ WoS,
                      const unsigned short* __restrict__ WoN,
                      const float*          __restrict__ bo,
                      unsigned short*       __restrict__ out_bf16,
                      float*                __restrict__ out_f32) {
    constexpr int MT  = 128;
    constexpr int ASH = D + 8;     // h buffer row stride (bf16)
    constexpr int CSs = D + 4;     // c buffer row stride (f32)
    constexpr int BS  = D + 8;     // staged-B row stride (bf16)
    constexpr int XS  = 72;        // staged-X row stride (bf16)
    constexpr int NT  = D / 16;    // column tiles per step
    constexpr int S   = DEG * NT;  // total tiles

    extern __shared__ char smem[];
    unsigned short* Ah   = (unsigned short*)smem;                               // [MT][ASH]
    float*          Cs   = (float*)(smem + (size_t)MT * ASH * 2);               // [MT][CSs]
    unsigned short* Bst0 = (unsigned short*)((char*)Cs + (size_t)MT * CSs * 4); // [64][BS]
    unsigned short* Bst1 = Bst0 + 64 * BS;
    unsigned short* Xst0 = Bst1 + 64 * BS;                                      // [MT][XS]
    unsigned short* Xst1 = Xst0 + MT * XS;

    const int tid  = threadIdx.x;
    const int lane = tid & 31;
    const int rowBase = (tid >> 5) << 4;        // wave's 16 rows
    const int base = blockIdx.x * MT;
    const int n16 = lane & 15;
    const int mhi = (lane >> 4) << 3;

    // staging decomposition (constant per thread)
    const int br = tid >> 2;                    // B: row 0..63
    const int bq = tid & 3;                     // B: quarter of a row
    const int xg = tid & 3;                     // X: gate
    const int xr0 = tid >> 2;                   // X: row (first chunk)

    // init h = 0, c = 0
    for (int i = tid; i < MT * D; i += 256) {
        const int r = i / D, cidx = i - r * D;
        Ah[r * ASH + cidx] = 0;
        Cs[r * CSs + cidx] = 0.0f;
    }

    // stage tile s into (Bdst, Xdst): W_hh rows {g*D + nt*16 + j} and gathered
    // Xproj pre-activation columns [g*D + nt*16, +16) for all MT rows.
    auto stage = [&](int s, unsigned short* Bdst, unsigned short* Xdst) {
        const int t  = s / NT;
        const int nt = s - t * NT;
        {
            const int wrow = (br >> 4) * D + nt * 16 + (br & 15);
            const unsigned short* src = Whh + (size_t)wrow * D + bq * (D / 4);
            unsigned short*       dst = Bdst + br * BS + bq * (D / 4);
#pragma unroll
            for (int j = 0; j < D / 32; ++j)
                async_copy_b128(dst + j * 8, src + j * 8);
        }
        {
            const int nb0 = neighbors[(base + xr0) * DEG + t];
            const unsigned short* src = Xproj + (size_t)nb0 * (4 * D) + xg * D + nt * 16;
            unsigned short*       dst = Xdst + xr0 * XS + xg * 16;
            async_copy_b128(dst, src);
            async_copy_b128(dst + 8, src + 8);
        }
        {
            const int xr1 = xr0 + 64;
            const int nb1 = neighbors[(base + xr1) * DEG + t];
            const unsigned short* src = Xproj + (size_t)nb1 * (4 * D) + xg * D + nt * 16;
            unsigned short*       dst = Xdst + xr1 * XS + xg * 16;
            async_copy_b128(dst, src);
            async_copy_b128(dst + 8, src + 8);
        }
    };

    // prologue: stage tile 0
    stage(0, Bst0, Xst0);
    async_wait();
    __syncthreads();

    // ------------------- flattened (step, tile) pipeline ------------------
    for (int s = 0; s < S; ++s) {
        unsigned short* Bcur = (s & 1) ? Bst1 : Bst0;
        unsigned short* Xcur = (s & 1) ? Xst1 : Xst0;
        if (s + 1 < S)
            stage(s + 1, (s & 1) ? Bst0 : Bst1, (s & 1) ? Xst0 : Xst1);

        const int nt = s - (s / NT) * NT;

        // gates for this 16-column tile: h @ Whh^T (h rows are wave-private)
        v8f ai = {}; v8f af = {}; v8f ag = {}; v8f ao = {};
#pragma unroll
        for (int kk = 0; kk < D / 32; ++kk) {
            const int kbase = kk * 32;
            FragB16 fa;
            load_a_tile<ASH>(Ah, rowBase, lane, kbase, fa);
            FragB16 f0, f1, f2, f3;
            load_b_tile_s(Bcur, BS,  0, lane, kbase, f0);
            load_b_tile_s(Bcur, BS, 16, lane, kbase, f1);
            load_b_tile_s(Bcur, BS, 32, lane, kbase, f2);
            load_b_tile_s(Bcur, BS, 48, lane, kbase, f3);
            ai = wmma_bf16(fa.v, f0.v, ai);
            af = wmma_bf16(fa.v, f1.v, af);
            ag = wmma_bf16(fa.v, f2.v, ag);
            ao = wmma_bf16(fa.v, f3.v, ao);
        }
        const int ncol = nt * 16 + n16;
#pragma unroll
        for (int v = 0; v < 8; ++v) {
            const int row = rowBase + v + mhi;
            const unsigned short* xr = Xcur + row * XS;
            const float iv = sigmoidf_fast(ai[v] + bf16_to_f32(xr[ 0 + n16]));
            const float fv = sigmoidf_fast(af[v] + bf16_to_f32(xr[16 + n16]));
            const float gv = tanhf       (ag[v] + bf16_to_f32(xr[32 + n16]));
            const float ov = sigmoidf_fast(ao[v] + bf16_to_f32(xr[48 + n16]));
            const float c  = fv * Cs[row * CSs + ncol] + iv * gv;
            Cs[row * CSs + ncol] = c;
            Ah[row * ASH + ncol] = f32_to_bf16(ov * tanhf(c));
        }

        async_wait();      // next tile staged (this wave)
        __syncthreads();   // next tile staged (all waves) + current buffers free
    }

    // ------------- output projection: feat@WoS + h@WoN + bo ---------------
    const int m = lane & 15;
    const int koffA = (lane >> 4) << 3;
    for (int nt = 0; nt < 256 / 16; ++nt) {
        v8f acc = {};
        const int cb = nt * 16;
#pragma unroll
        for (int kk = 0; kk < D / 32; ++kk) {          // self part, A from global
            const int kbase = kk * 32;
            FragB16 fa, fb;
            const unsigned short* p = featprev + (size_t)(base + rowBase + m) * D + kbase + koffA;
            fa.u[0] = *(const uint4*)(p);
            fa.u[1] = *(const uint4*)(p + 16);
            load_b_tile_s(WoS, D, cb, lane, kbase, fb);
            acc = wmma_bf16(fa.v, fb.v, acc);
        }
#pragma unroll
        for (int kk = 0; kk < D / 32; ++kk) {          // neighbor part, A = h in LDS
            const int kbase = kk * 32;
            FragB16 fa, fb;
            load_a_tile<ASH>(Ah, rowBase, lane, kbase, fa);
            load_b_tile_s(WoN, D, cb, lane, kbase, fb);
            acc = wmma_bf16(fa.v, fb.v, acc);
        }
        const int ncolO = cb + n16;
        const float bb = bo[ncolO];
#pragma unroll
        for (int v = 0; v < 8; ++v) {
            const int node = base + rowBase + v + mhi;
            float val = acc[v] + bb;
            if (RELU_BF16_OUT) {
                val = fmaxf(val, 0.0f);
                out_bf16[(size_t)node * 256 + ncolO] = f32_to_bf16(val);
            } else {
                out_f32[(size_t)node * 256 + ncolO] = val;
            }
        }
    }
}

// ---------------------------------------------------------------------------
// Per-graph mean over sorted node2graph (binary search, no atomics).
// ---------------------------------------------------------------------------
__global__ __launch_bounds__(256)
void pool_mean_kernel(const float* __restrict__ h2,
                      const int*   __restrict__ node2graph,
                      float*       __restrict__ pooled) {
    const int g = blockIdx.x;
    const int col = threadIdx.x;
    int lo = 0, hi = NN;
    while (lo < hi) { int mm = (lo + hi) >> 1; if (node2graph[mm] <  g) lo = mm + 1; else hi = mm; }
    const int start = lo;
    lo = start; hi = NN;
    while (lo < hi) { int mm = (lo + hi) >> 1; if (node2graph[mm] <= g) lo = mm + 1; else hi = mm; }
    const int end = lo;
    float s = 0.0f;
    for (int i = start; i < end; ++i) s += h2[(size_t)i * 256 + col];
    pooled[g * 256 + col] = s / fmaxf((float)(end - start), 1.0f);
}

// ---------------------------------------------------------------------------
// Heads: tiny (64x256x128) scalar GEMMs.
// ---------------------------------------------------------------------------
__global__ __launch_bounds__(128)
void heads_kernel(const float* __restrict__ pooled,
                  const float* __restrict__ w_mu,  const float* __restrict__ b_mu,
                  const float* __restrict__ w_sig, const float* __restrict__ b_sig,
                  float* __restrict__ out) {
    const int g = blockIdx.x;
    const int n = threadIdx.x;
    float sm = 0.0f, ss = 0.0f;
    for (int k = 0; k < 256; ++k) {
        const float x = pooled[g * 256 + k];
        sm = fmaf(x, w_mu[k * 128 + n],  sm);
        ss = fmaf(x, w_sig[k * 128 + n], ss);
    }
    out[g * 128 + n]            = sm + b_mu[n];
    out[GG * 128 + g * 128 + n] = ss + b_sig[n];
}

// ---------------------------------------------------------------------------
// Prep kernels.
// ---------------------------------------------------------------------------
__global__ void cvt_bf16_kernel(const float* __restrict__ src,
                                unsigned short* __restrict__ dst, int n) {
    const int i = blockIdx.x * 256 + threadIdx.x;
    if (i < n) dst[i] = f32_to_bf16(src[i]);
}

// dst[n][k] = w[k*dout + n]  (transpose-pack, dst is [dout][d] bf16)
__global__ void tpack_kernel(const float* __restrict__ w,
                             unsigned short* __restrict__ dst, int dout, int d) {
    const int i = blockIdx.x * 256 + threadIdx.x;
    if (i >= dout * d) return;
    const int n = i / d, k = i - n * d;
    dst[i] = f32_to_bf16(w[k * dout + n]);
}

// ---------------------------------------------------------------------------
extern "C" void kernel_launch(void* const* d_in, const int* in_sizes, int n_in,
                              void* d_out, int out_size, void* d_ws, size_t ws_size,
                              hipStream_t stream) {
    (void)in_sizes; (void)n_in; (void)out_size; (void)ws_size;

    const float* in_feat   = (const float*)d_in[0];
    const int*   neighbors = (const int*)  d_in[1];
    const int*   node2g    = (const int*)  d_in[2];
    const float* w_ih1     = (const float*)d_in[3];
    const float* w_hh1     = (const float*)d_in[4];
    const float* b_lstm1   = (const float*)d_in[5];
    const float* w_self1   = (const float*)d_in[6];
    const float* w_neigh1  = (const float*)d_in[7];
    const float* b1        = (const float*)d_in[8];
    const float* w_ih2     = (const float*)d_in[9];
    const float* w_hh2     = (const float*)d_in[10];
    const float* b_lstm2   = (const float*)d_in[11];
    const float* w_self2   = (const float*)d_in[12];
    const float* w_neigh2  = (const float*)d_in[13];
    const float* b2        = (const float*)d_in[14];
    const float* w_mu      = (const float*)d_in[15];
    const float* b_mu      = (const float*)d_in[16];
    const float* w_sigma   = (const float*)d_in[17];
    const float* b_sigma   = (const float*)d_in[18];
    float* out = (float*)d_out;

    char* ws = (char*)d_ws;
    unsigned short* feat_bf = (unsigned short*)(ws);               // N*128  bf16
    unsigned short* h1_bf   = (unsigned short*)(ws +  8388608);    // N*256  bf16
    unsigned short* Wih1    = (unsigned short*)(ws + 25165824);    // 512*128
    unsigned short* Whh1    = (unsigned short*)(ws + 25296896);    // 512*128
    unsigned short* Wih2    = (unsigned short*)(ws + 25427968);    // 1024*256
    unsigned short* Whh2    = (unsigned short*)(ws + 25952256);    // 1024*256
    unsigned short* WoS1    = (unsigned short*)(ws + 26476544);    // 256*128
    unsigned short* WoN1    = (unsigned short*)(ws + 26542080);    // 256*128
    unsigned short* WoS2    = (unsigned short*)(ws + 26607616);    // 256*256
    unsigned short* WoN2    = (unsigned short*)(ws + 26738688);    // 256*256
    float*          pooled  = (float*)         (ws + 26869760);    // 64*256
    unsigned short* Xproj   = (unsigned short*)(ws + 27262976);    // N*1024 bf16 (reused)
    float*          h2_f32  = (float*)         (ws + 94371840);    // N*256  f32

    // ---- weight/activation prep ----
    cvt_bf16_kernel<<<(NN * 128  + 255) / 256, 256, 0, stream>>>(in_feat, feat_bf, NN * 128);
    cvt_bf16_kernel<<<(512 * 128 + 255) / 256, 256, 0, stream>>>(w_ih1, Wih1, 512 * 128);
    cvt_bf16_kernel<<<(512 * 128 + 255) / 256, 256, 0, stream>>>(w_hh1, Whh1, 512 * 128);
    cvt_bf16_kernel<<<(1024 * 256 + 255) / 256, 256, 0, stream>>>(w_ih2, Wih2, 1024 * 256);
    cvt_bf16_kernel<<<(1024 * 256 + 255) / 256, 256, 0, stream>>>(w_hh2, Whh2, 1024 * 256);
    tpack_kernel<<<(256 * 128 + 255) / 256, 256, 0, stream>>>(w_self1,  WoS1, 256, 128);
    tpack_kernel<<<(256 * 128 + 255) / 256, 256, 0, stream>>>(w_neigh1, WoN1, 256, 128);
    tpack_kernel<<<(256 * 256 + 255) / 256, 256, 0, stream>>>(w_self2,  WoS2, 256, 256);
    tpack_kernel<<<(256 * 256 + 255) / 256, 256, 0, stream>>>(w_neigh2, WoN2, 256, 256);

    // ---- dynamic LDS sizes (double-buffered staging) ----
    constexpr unsigned lds1 = 128*(128+8)*2 + 128*(128+4)*4 + 2*64*(128+8)*2 + 2*128*72*2; // 174080
    constexpr unsigned lds2 = 128*(256+8)*2 + 128*(256+4)*4 + 2*64*(256+8)*2 + 2*128*72*2; // 305152
    (void)hipFuncSetAttribute(reinterpret_cast<const void*>(&sage_lstm_kernel<128, true>),
                              hipFuncAttributeMaxDynamicSharedMemorySize, (int)lds1);
    (void)hipFuncSetAttribute(reinterpret_cast<const void*>(&sage_lstm_kernel<256, false>),
                              hipFuncAttributeMaxDynamicSharedMemorySize, (int)lds2);

    // ---- layer 1 ----
    xproj_kernel<128><<<NN / 64, 128, 0, stream>>>(feat_bf, Wih1, b_lstm1, Xproj);
    sage_lstm_kernel<128, true><<<NN / 128, 256, lds1, stream>>>(
        Xproj, feat_bf, neighbors, Whh1, WoS1, WoN1, b1, h1_bf, nullptr);

    // ---- layer 2 (Xproj buffer reused) ----
    xproj_kernel<256><<<NN / 64, 128, 0, stream>>>(h1_bf, Wih2, b_lstm2, Xproj);
    sage_lstm_kernel<256, false><<<NN / 128, 256, lds2, stream>>>(
        Xproj, h1_bf, neighbors, Whh2, WoS2, WoN2, b2, nullptr, h2_f32);

    // ---- pooling + heads ----
    pool_mean_kernel<<<GG, 256, 0, stream>>>(h2_f32, node2g, pooled);
    heads_kernel<<<GG, 128, 0, stream>>>(pooled, w_mu, b_mu, w_sigma, b_sigma, out);
}
